// HeatmapRegressionHead_63050119905590
// MI455X (gfx1250) — compile-verified, matrix operand order checked by reading, not run on hardware
//
#include <hip/hip_runtime.h>
#include <hip/hip_bf16.h>

typedef __attribute__((ext_vector_type(16))) __bf16 bf16x16;
typedef __attribute__((ext_vector_type(8)))  float  f32x8;
typedef __attribute__((ext_vector_type(4)))  unsigned int u32x4;
typedef __attribute__((ext_vector_type(8)))  unsigned int u32x8;

#define DEVINL __device__ __forceinline__

static constexpr int BATCH = 32;
static constexpr int HH    = 64;
static constexpr int WWID  = 64;
static constexpr int HW    = HH * WWID;       // 4096
static constexpr int CINC  = 256;             // every conv has Cin==256 (v2 head is 128)
static constexpr int KPTS  = 17;

// d_out layout (floats): heat | off | var | fw | coords | scores
static constexpr size_t OFF_HEAT   = 0;
static constexpr size_t SZ_HEAT    = (size_t)BATCH * KPTS * HW;        // 2228224
static constexpr size_t OFF_OFF    = OFF_HEAT + SZ_HEAT;
static constexpr size_t SZ_OFF     = (size_t)BATCH * 2 * KPTS * HW;    // 4456448
static constexpr size_t OFF_VAR    = OFF_OFF + SZ_OFF;
static constexpr size_t SZ_VAR     = SZ_HEAT;
static constexpr size_t OFF_FW     = OFF_VAR + SZ_VAR;
static constexpr size_t OFF_COORDS = OFF_FW + 1;
static constexpr size_t OFF_SCORES = OFF_COORDS + (size_t)BATCH * KPTS * 2;

DEVINL unsigned short f2bf(float f) {
  unsigned u = __builtin_bit_cast(unsigned, f);
  unsigned r = 0x7FFFu + ((u >> 16) & 1u);
  return (unsigned short)((u + r) >> 16);
}

// ---------------- layout / weight prep kernels ----------------

__global__ void nchw2nhwc_bf16(const float* __restrict__ src,
                               unsigned short* __restrict__ dst,
                               long long total, int C) {
  long long i = (long long)blockIdx.x * 256 + threadIdx.x;
  if (i >= total) return;
  long long b   = i / ((long long)C * HW);
  long long rem = i - b * (long long)C * HW;
  int c = (int)(rem / HW);
  int p = (int)(rem - (long long)c * HW);
  dst[((size_t)b * HW + p) * C + c] = f2bf(src[i]);
}

// w [Cout][Cin][3][3] f32  ->  wb [9][Cout][Cin] bf16
__global__ void prep_w3x3(const float* __restrict__ src,
                          unsigned short* __restrict__ dst, int Cout, int Cin) {
  long long total = (long long)Cout * Cin * 9;
  long long i = (long long)blockIdx.x * 256 + threadIdx.x;
  if (i >= total) return;
  int t = (int)(i % 9);
  long long coci = i / 9;
  int ci = (int)(coci % Cin);
  int co = (int)(coci / Cin);
  dst[((size_t)t * Cout + co) * Cin + ci] = f2bf(src[i]);
}

// w [Cout][Cin] f32 -> wb [Mpad][Cin] bf16 (zero padded rows)
__global__ void prep_w1x1(const float* __restrict__ src,
                          unsigned short* __restrict__ dst,
                          int Cout, int Cin, int Mpad) {
  long long total = (long long)Mpad * Cin;
  long long i = (long long)blockIdx.x * 256 + threadIdx.x;
  if (i >= total) return;
  int m  = (int)(i / Cin);
  int ci = (int)(i % Cin);
  dst[i] = (m < Cout) ? f2bf(src[(size_t)m * Cin + ci]) : (unsigned short)0;
}

// ---------------- 3x3 conv, implicit GEMM via WMMA ----------------
// act: bf16 NHWC [B*HW][256]; wgt: bf16 [9][Cout][256]; out: bf16 NHWC [B*HW][Cout]
// grid: (B*H = 2048, Cout/64); block: 256 (8 waves); waves = 4(M) x 2(N); wave tile 16x32

__global__ __launch_bounds__(256) void conv3x3_bf16_wmma(
    const unsigned short* __restrict__ act,
    const unsigned short* __restrict__ wgt,
    const float* __restrict__ gam, const float* __restrict__ bet,
    unsigned short* __restrict__ outp, int Cout) {
  __shared__ __align__(16) unsigned short lds_w[64 * CINC];  // 32 KB weight tile (TDM)
  __shared__ __align__(16) unsigned short lds_a[64 * CINC];  // 32 KB activation tile (async)

  const int tid  = threadIdx.x;
  const int wave = tid >> 5, lane = tid & 31;
  const int wm = wave >> 1, wn = wave & 1;
  const int m0 = blockIdx.y * 64;
  const int bb = (int)blockIdx.x >> 6;   // batch
  const int hh = (int)blockIdx.x & 63;   // row
  const int halfk = lane & 16;

  const int mrow  = wm * 16 + (lane & 15);
  const int nrow0 = wn * 32 + (lane & 15);
  const int nrow1 = nrow0 + 16;

  f32x8 acc0 = {0, 0, 0, 0, 0, 0, 0, 0};
  f32x8 acc1 = {0, 0, 0, 0, 0, 0, 0, 0};

  const int nl    = tid & 63;   // spatial slot this thread stages
  const int cpart = tid >> 6;   // 64-channel slice (0..3)
  const unsigned la_base =
      (unsigned)(size_t)(const void*)&lds_a[nl * CINC + cpart * 64];

#pragma unroll 1
  for (int tap = 0; tap < 9; ++tap) {
    const int dy = tap / 3 - 1, dx = tap % 3 - 1;
    __syncthreads();  // previous tap's LDS reads complete

    if (tid < 32) {
      // Tensor Data Mover: 2D tile [64 rows][256 cols] of bf16 weights -> lds_w
      unsigned long long ga =
          (unsigned long long)(size_t)(wgt + ((size_t)tap * Cout + m0) * CINC);
      unsigned ldsw = (unsigned)(size_t)(const void*)&lds_w[0];
      u32x4 g0; u32x8 g1;
      g0[0] = 1u;                                              // count=1, user mode
      g0[1] = ldsw;                                            // lds_addr
      g0[2] = (unsigned)ga;                                    // global_addr lo
      g0[3] = (unsigned)((ga >> 32) & 0x01FFFFFFull) | (2u << 30);  // addr hi | type=2
      g1[0] = (1u << 16);                                      // data_size = 2 bytes
      g1[1] = ((unsigned)CINC & 0xFFFFu) << 16;                // tensor_dim0 lo16
      g1[2] = (((unsigned)CINC >> 16) & 0xFFFFu) |
              (((unsigned)Cout & 0xFFFFu) << 16);              // dim0 hi | dim1 lo
      g1[3] = (((unsigned)Cout >> 16) & 0xFFFFu) |
              ((unsigned)CINC << 16);                          // dim1 hi | tile_dim0
      g1[4] = 64u;                                             // tile_dim1=64, tile_dim2=0
      g1[5] = (unsigned)CINC;                                  // tensor_dim0_stride lo
      g1[6] = 0u; g1[7] = 0u;
      asm volatile("tensor_load_to_lds %0, %1" ::"s"(g0), "s"(g1) : "memory");
    }

    {  // async global->LDS: one row (clamped), w shifted by dx, all 256 channels
      int hp = hh + dy; hp = hp < 0 ? 0 : (hp > 63 ? 63 : hp);
      int wp = nl + dx; wp = wp < 0 ? 0 : (wp > 63 ? 63 : wp);
      unsigned long long ga = (unsigned long long)(size_t)(
          act + ((size_t)((bb * 64 + hp) * 64 + wp)) * CINC + cpart * 64);
#pragma unroll
      for (int j = 0; j < 8; ++j) {
        asm volatile("global_load_async_to_lds_b128 %0, %1, off" ::
                         "v"(la_base + (unsigned)(j * 16)),
                         "v"(ga + (unsigned long long)(j * 16))
                     : "memory");
      }
    }

    asm volatile("s_wait_asynccnt 0x0" ::: "memory");
    __builtin_amdgcn_s_wait_tensorcnt(0);
    __syncthreads();

#pragma unroll
    for (int kc = 0; kc < 8; ++kc) {
      const int kb = kc * 32 + halfk;
      bf16x16 afr = *reinterpret_cast<const bf16x16*>(&lds_w[mrow  * CINC + kb]);
      bf16x16 bf0 = *reinterpret_cast<const bf16x16*>(&lds_a[nrow0 * CINC + kb]);
      bf16x16 bf1 = *reinterpret_cast<const bf16x16*>(&lds_a[nrow1 * CINC + kb]);
      acc0 = __builtin_amdgcn_wmma_f32_16x16x32_bf16(false, afr, false, bf0,
                                                     (short)0, acc0, false, false);
      acc1 = __builtin_amdgcn_wmma_f32_16x16x32_bf16(false, afr, false, bf1,
                                                     (short)0, acc1, false, false);
    }
  }

  // fused BN + ReLU epilogue -> bf16 NHWC
  const float rs = rsqrtf(1.0f + 1e-5f);
  const int mbase = m0 + wm * 16 + ((lane >> 4) << 3);
  const size_t n0g = (size_t)blockIdx.x * 64 + wn * 32 + (lane & 15);
  const size_t n1g = n0g + 16;
#pragma unroll
  for (int r = 0; r < 8; ++r) {
    const int m = mbase + r;
    const float sc = gam[m] * rs;
    const float bi = bet[m];
    outp[n0g * Cout + m] = f2bf(fmaxf(fmaf(acc0[r], sc, bi), 0.0f));
    outp[n1g * Cout + m] = f2bf(fmaxf(fmaf(acc1[r], sc, bi), 0.0f));
  }
}

// ---------------- 1x1 head GEMM via WMMA (bias + optional softplus) ----------------
// grid: (2048, Mpad/32); block 256; waves = 2(M) x 4(N); wave tile 16x16

__global__ __launch_bounds__(256) void gemm1x1_wmma(
    const unsigned short* __restrict__ act,  // [B*HW][Cin] bf16
    const unsigned short* __restrict__ wgt,  // [Mpad][Cin] bf16
    const float* __restrict__ bias,
    float* __restrict__ outp,                // f32 NCHW [B][CoutA][HW]
    int Cin, int CoutA, int actcode) {
  const int tid  = threadIdx.x;
  const int wave = tid >> 5, lane = tid & 31;
  const int wm = wave >> 2, wnn = wave & 3;
  const int halfk = lane & 16;
  const int m0 = blockIdx.y * 32 + wm * 16;
  const size_t n = (size_t)blockIdx.x * 64 + wnn * 16 + (lane & 15);
  const int mrow = m0 + (lane & 15);

  f32x8 acc = {0, 0, 0, 0, 0, 0, 0, 0};
  for (int kb = 0; kb < Cin; kb += 32) {
    bf16x16 afr = *reinterpret_cast<const bf16x16*>(&wgt[(size_t)mrow * Cin + kb + halfk]);
    bf16x16 bfr = *reinterpret_cast<const bf16x16*>(&act[n * Cin + kb + halfk]);
    acc = __builtin_amdgcn_wmma_f32_16x16x32_bf16(false, afr, false, bfr,
                                                  (short)0, acc, false, false);
  }
  const int mbase = m0 + ((lane >> 4) << 3);
  const size_t b = n >> 12;
  const size_t p = n & 4095;
#pragma unroll
  for (int r = 0; r < 8; ++r) {
    const int m = mbase + r;
    if (m < CoutA) {
      float v = acc[r] + bias[m];
      if (actcode == 1) v = (v > 20.0f) ? v : log1pf(__expf(v));
      outp[((size_t)b * CoutA + m) * HW + p] = v;
    }
  }
}

// ---------------- soft-argmax + refine + offset sampling ----------------
// one block per (b,k)

__global__ __launch_bounds__(256) void heads_post(
    const float* __restrict__ heat, const float* __restrict__ off,
    float* __restrict__ coords, float* __restrict__ scores,
    float* __restrict__ fwout, const float* __restrict__ alpha_p,
    const float* __restrict__ fusion_p) {
  const int bk = blockIdx.x;
  const float* hm = heat + (size_t)bk * HW;
  __shared__ float red[256];
  const int tid = threadIdx.x;

  float lm = -3.4e38f;
  for (int i = tid; i < HW; i += 256) lm = fmaxf(lm, hm[i]);
  red[tid] = lm; __syncthreads();
  for (int s = 128; s > 0; s >>= 1) {
    if (tid < s) red[tid] = fmaxf(red[tid], red[tid + s]);
    __syncthreads();
  }
  const float gmax = red[0]; __syncthreads();

  float se = 0.f, sx = 0.f, sy = 0.f;
  for (int i = tid; i < HW; i += 256) {
    float e = __expf(hm[i] - gmax);
    se += e; sx += e * (float)(i & 63); sy += e * (float)(i >> 6);
  }
  red[tid] = se; __syncthreads();
  for (int s = 128; s > 0; s >>= 1) { if (tid < s) red[tid] += red[tid + s]; __syncthreads(); }
  se = red[0]; __syncthreads();
  red[tid] = sx; __syncthreads();
  for (int s = 128; s > 0; s >>= 1) { if (tid < s) red[tid] += red[tid + s]; __syncthreads(); }
  sx = red[0]; __syncthreads();
  red[tid] = sy; __syncthreads();
  for (int s = 128; s > 0; s >>= 1) { if (tid < s) red[tid] += red[tid + s]; __syncthreads(); }
  sy = red[0];

  if (tid == 0) {
    const float cgx = sx / se, cgy = sy / se;
    // local 5x5 refine
    int px = (int)roundf(fminf(fmaxf(cgx, 0.f), 63.f));
    int py = (int)roundf(fminf(fmaxf(cgy, 0.f), 63.f));
    float pmax = -3.4e38f;
    for (int j = -2; j <= 2; ++j)
      for (int i = -2; i <= 2; ++i) {
        int ys = py + j, xs = px + i;
        if (ys >= 0 && ys < 64 && xs >= 0 && xs < 64)
          pmax = fmaxf(pmax, hm[ys * 64 + xs]);
      }
    float wsum = 0.f, rx = 0.f, ry = 0.f;
    for (int j = -2; j <= 2; ++j)
      for (int i = -2; i <= 2; ++i) {
        int ys = py + j, xs = px + i;
        int yc = min(max(ys, 0), 63), xc = min(max(xs, 0), 63);
        bool ok = (ys >= 0 && ys < 64 && xs >= 0 && xs < 64);
        float e = ok ? __expf(hm[yc * 64 + xc] - pmax) : 0.f;
        wsum += e; rx += e * (float)xc; ry += e * (float)yc;
      }
    rx /= wsum; ry /= wsum;

    const float a  = 1.f / (1.f + __expf(-alpha_p[0]));
    const float fw = 1.f / (1.f + __expf(-fusion_p[0]));
    float cx = a * cgx + (1.f - a) * rx;
    float cy = a * cgy + (1.f - a) * ry;

    // bilinear sample of offset field
    float ix = fminf(fmaxf(cx, 0.f), 63.f), iy = fminf(fmaxf(cy, 0.f), 63.f);
    float x0 = floorf(ix), y0 = floorf(iy);
    float wx = ix - x0, wy = iy - y0;
    int x0i = min(max((int)x0, 0), 63), x1i = min(x0i + 1, 63);
    int y0i = min(max((int)y0, 0), 63), y1i = min(y0i + 1, 63);
#pragma unroll
    for (int c = 0; c < 2; ++c) {
      const float* oc = off + ((size_t)bk * 2 + c) * HW;
      float v00 = oc[y0i * 64 + x0i], v01 = oc[y0i * 64 + x1i];
      float v10 = oc[y1i * 64 + x0i], v11 = oc[y1i * 64 + x1i];
      float v = (1.f - wy) * ((1.f - wx) * v00 + wx * v01) +
                wy * ((1.f - wx) * v10 + wx * v11);
      if (c == 0) cx += fw * v; else cy += fw * v;
    }
    coords[(size_t)bk * 2 + 0] = cx;
    coords[(size_t)bk * 2 + 1] = cy;
    scores[bk] = gmax;
    if (bk == 0) fwout[0] = fw;
  }
}

// ---------------- host launcher ----------------

extern "C" void kernel_launch(void* const* d_in, const int* in_sizes, int n_in,
                              void* d_out, int out_size, void* d_ws, size_t ws_size,
                              hipStream_t stream) {
  const float* x    = (const float*)d_in[0];
  const float* w_s1 = (const float*)d_in[1];
  const float* g_s1 = (const float*)d_in[2];
  const float* b_s1 = (const float*)d_in[3];
  const float* w_s2 = (const float*)d_in[4];
  const float* g_s2 = (const float*)d_in[5];
  const float* b_s2 = (const float*)d_in[6];
  const float* w_h1 = (const float*)d_in[7];
  const float* g_h1 = (const float*)d_in[8];
  const float* b_h1 = (const float*)d_in[9];
  const float* w_h2 = (const float*)d_in[10];
  const float* c_h2 = (const float*)d_in[11];
  const float* w_o1 = (const float*)d_in[12];
  const float* g_o1 = (const float*)d_in[13];
  const float* b_o1 = (const float*)d_in[14];
  const float* w_o2 = (const float*)d_in[15];
  const float* c_o2 = (const float*)d_in[16];
  const float* w_v1 = (const float*)d_in[17];
  const float* g_v1 = (const float*)d_in[18];
  const float* b_v1 = (const float*)d_in[19];
  const float* w_v2 = (const float*)d_in[20];
  const float* c_v2 = (const float*)d_in[21];
  const float* alpha  = (const float*)d_in[22];
  const float* fusion = (const float*)d_in[23];
  float* out = (float*)d_out;

  char* ws = (char*)d_ws;
  size_t cur = 0;
  auto alloc = [&](size_t bytes) -> char* {
    char* p = ws + cur;
    cur += (bytes + 255) & ~(size_t)255;
    return p;
  };
  const size_t actBytes = (size_t)BATCH * HW * CINC * 2;          // 64 MB
  unsigned short* buf0 = (unsigned short*)alloc(actBytes);
  unsigned short* buf1 = (unsigned short*)alloc(actBytes);
  unsigned short* buf2 = (unsigned short*)alloc(actBytes);
  unsigned short* wb_s1 = (unsigned short*)alloc((size_t)9 * 256 * 256 * 2);
  unsigned short* wb_s2 = (unsigned short*)alloc((size_t)9 * 256 * 256 * 2);
  unsigned short* wb_h1 = (unsigned short*)alloc((size_t)9 * 256 * 256 * 2);
  unsigned short* wb_o1 = (unsigned short*)alloc((size_t)9 * 256 * 256 * 2);
  unsigned short* wb_v1 = (unsigned short*)alloc((size_t)9 * 128 * 256 * 2);
  unsigned short* wb_h2 = (unsigned short*)alloc((size_t)32 * 256 * 2);
  unsigned short* wb_o2 = (unsigned short*)alloc((size_t)64 * 256 * 2);
  unsigned short* wb_v2 = (unsigned short*)alloc((size_t)32 * 128 * 2);

  dim3 blk(256);
  {
    long long n = (long long)BATCH * CINC * HW;
    nchw2nhwc_bf16<<<dim3((unsigned)((n + 255) / 256)), blk, 0, stream>>>(x, buf0, n, CINC);
  }
  {
    long long n3 = (long long)9 * 256 * 256;
    unsigned g3 = (unsigned)((n3 + 255) / 256);
    prep_w3x3<<<g3, blk, 0, stream>>>(w_s1, wb_s1, 256, 256);
    prep_w3x3<<<g3, blk, 0, stream>>>(w_s2, wb_s2, 256, 256);
    prep_w3x3<<<g3, blk, 0, stream>>>(w_h1, wb_h1, 256, 256);
    prep_w3x3<<<g3, blk, 0, stream>>>(w_o1, wb_o1, 256, 256);
    long long nv = (long long)9 * 128 * 256;
    prep_w3x3<<<(unsigned)((nv + 255) / 256), blk, 0, stream>>>(w_v1, wb_v1, 128, 256);
    prep_w1x1<<<(32 * 256 + 255) / 256, blk, 0, stream>>>(w_h2, wb_h2, 17, 256, 32);
    prep_w1x1<<<(64 * 256 + 255) / 256, blk, 0, stream>>>(w_o2, wb_o2, 34, 256, 64);
    prep_w1x1<<<(32 * 128 + 255) / 256, blk, 0, stream>>>(w_v2, wb_v2, 17, 128, 32);
  }

  // backbone
  conv3x3_bf16_wmma<<<dim3(2048, 4), blk, 0, stream>>>(buf0, wb_s1, g_s1, b_s1, buf1, 256);
  conv3x3_bf16_wmma<<<dim3(2048, 4), blk, 0, stream>>>(buf1, wb_s2, g_s2, b_s2, buf2, 256);
  // heatmap head
  conv3x3_bf16_wmma<<<dim3(2048, 4), blk, 0, stream>>>(buf2, wb_h1, g_h1, b_h1, buf0, 256);
  gemm1x1_wmma<<<dim3(2048, 1), blk, 0, stream>>>(buf0, wb_h2, c_h2, out + OFF_HEAT, 256, 17, 0);
  // offset head
  conv3x3_bf16_wmma<<<dim3(2048, 4), blk, 0, stream>>>(buf2, wb_o1, g_o1, b_o1, buf0, 256);
  gemm1x1_wmma<<<dim3(2048, 2), blk, 0, stream>>>(buf0, wb_o2, c_o2, out + OFF_OFF, 256, 34, 0);
  // variance head (Cout=128 conv, softplus 1x1)
  conv3x3_bf16_wmma<<<dim3(2048, 2), blk, 0, stream>>>(buf2, wb_v1, g_v1, b_v1, buf0, 128);
  gemm1x1_wmma<<<dim3(2048, 1), blk, 0, stream>>>(buf0, wb_v2, c_v2, out + OFF_VAR, 128, 17, 1);

  // coords / scores / fw
  heads_post<<<dim3(BATCH * KPTS), blk, 0, stream>>>(
      out + OFF_HEAT, out + OFF_OFF, out + OFF_COORDS, out + OFF_SCORES,
      out + OFF_FW, alpha, fusion);

  (void)in_sizes; (void)n_in; (void)out_size; (void)ws_size;
}
